// CrossAttention_25795573579905
// MI455X (gfx1250) — matrix operand order repaired and checked by reading, not compiled, not run on hardware
//
#include <hip/hip_runtime.h>
#include <hip/hip_bf16.h>
#include <math.h>

// ---------------------------------------------------------------------------
// CDNA5 (gfx1250) cross-attention, bf16 WMMA + f32 accumulate, wave32.
// ---------------------------------------------------------------------------

typedef __bf16 v16bf __attribute__((ext_vector_type(16)));
typedef float  v8f   __attribute__((ext_vector_type(8)));

__device__ __forceinline__ v8f wmma_bf16(v16bf a, v16bf b, v8f c) {
    // D = A(16x32) * B(32x16) + C(16x16), f32 accum
    return __builtin_amdgcn_wmma_f32_16x16x32_bf16(
        false, a, false, b, (short)0, c, false, false);
}

// ---- fragment loaders (contiguous k order k = 16*(l/16) + s, same in A and B) -----------------
// A 16x32 bf16: lane m=l&15; lanes 0-15 K 0..15, lanes 16-31 K 16..31 (pairs)
// B 32x16 bf16: lane n=l&15; lanes 0-15 K 0..15, lanes 16-31 K 16..31 (pairs)

__device__ __forceinline__ v16bf a_frag_f32(const float* p, int ld) {
    const int lane = threadIdx.x & 31;
    const int m    = lane & 15;
    const int kb   = (lane & 16) ? 16 : 0;
    const float* rp = p + (size_t)m * ld;
    v16bf a;
#pragma unroll
    for (int vv = 0; vv < 8; ++vv) {
        const int kk = kb + 2 * vv;
        a[2 * vv]     = (__bf16)rp[kk];
        a[2 * vv + 1] = (__bf16)rp[kk + 1];
    }
    return a;
}

__device__ __forceinline__ v16bf a_frag_bf16(const __bf16* p, int ld) {
    const int lane = threadIdx.x & 31;
    const int m    = lane & 15;
    const int kb   = (lane & 16) ? 16 : 0;
    const __bf16* rp = p + (size_t)m * ld;
    v16bf a;
#pragma unroll
    for (int vv = 0; vv < 8; ++vv) {
        const int kk = kb + 2 * vv;
        a[2 * vv]     = rp[kk];
        a[2 * vv + 1] = rp[kk + 1];
    }
    return a;
}

// Same, but multiplies by a scalar during load (used to fold the softmax
// scale into Q, matching the reference's q = split_heads(q) * scale).
__device__ __forceinline__ v16bf a_frag_bf16_scaled(const __bf16* p, int ld, float s) {
    const int lane = threadIdx.x & 31;
    const int m    = lane & 15;
    const int kb   = (lane & 16) ? 16 : 0;
    const __bf16* rp = p + (size_t)m * ld;
    v16bf a;
#pragma unroll
    for (int vv = 0; vv < 8; ++vv) {
        const int kk = kb + 2 * vv;
        a[2 * vv]     = (__bf16)((float)rp[kk] * s);
        a[2 * vv + 1] = (__bf16)((float)rp[kk + 1] * s);
    }
    return a;
}

// B element (k,n) at p[k*ld + n]  (row-major K x N, f32 source)
__device__ __forceinline__ v16bf b_frag_f32_krows(const float* p, int ld) {
    const int lane = threadIdx.x & 31;
    const int n    = lane & 15;
    const int kb   = (lane & 16) ? 16 : 0;
    const float* cp = p + n + (size_t)kb * ld;
    v16bf b;
#pragma unroll
    for (int vv = 0; vv < 8; ++vv) {
        b[2 * vv]     = (__bf16)cp[(size_t)(2 * vv) * ld];
        b[2 * vv + 1] = (__bf16)cp[(size_t)(2 * vv + 1) * ld];
    }
    return b;
}

// B element (k,n) at p[k*ld + n]  (row-major K x N, bf16 source)
__device__ __forceinline__ v16bf b_frag_bf16_krows(const __bf16* p, int ld) {
    const int lane = threadIdx.x & 31;
    const int n    = lane & 15;
    const int kb   = (lane & 16) ? 16 : 0;
    const __bf16* cp = p + n + (size_t)kb * ld;
    v16bf b;
#pragma unroll
    for (int vv = 0; vv < 8; ++vv) {
        b[2 * vv]     = cp[(size_t)(2 * vv) * ld];
        b[2 * vv + 1] = cp[(size_t)(2 * vv + 1) * ld];
    }
    return b;
}

// B element (k,n) at p[n*ld + k]  ("transposed" source: K pairs contiguous,
// lowers to two global_load_b128 per fragment)
__device__ __forceinline__ v16bf b_frag_bf16_ncols(const __bf16* p, int ld) {
    const int lane = threadIdx.x & 31;
    const int n    = lane & 15;
    const int kb   = (lane & 16) ? 16 : 0;
    const __bf16* cp = p + (size_t)n * ld + kb;
    v16bf b;
#pragma unroll
    for (int vv = 0; vv < 8; ++vv) {
        b[2 * vv]     = cp[2 * vv];
        b[2 * vv + 1] = cp[2 * vv + 1];
    }
    return b;
}

// ---------------------------------------------------------------------------
// Kernel 1: QKV projection.  out[o][n] = sum_c W[o][c] * in[c][n] + bias[o]
// MODE 0: store bf16 as [b][h][n][d]  (q, k)     MODE 2: [b][h][d][n]  (v)
// Block = 256 thr = 8 waves; wave tile 32(o) x 64(n); block tile 64 x 256.
// Grid (4 n-tiles, 8 o-tiles, 8 batches).
// ---------------------------------------------------------------------------
template <int MODE>
__global__ __launch_bounds__(256) void proj_qkv_kernel(
    const float* __restrict__ in,    // [b][512][1024]
    const float* __restrict__ W,     // [512][512]
    const float* __restrict__ bias,  // [512]
    __bf16* __restrict__ out)
{
    const int lane = threadIdx.x & 31;
    const int wave = threadIdx.x >> 5;
    const int b    = blockIdx.z;
    const int o0   = blockIdx.y * 64 + (wave & 1) * 32;
    const int n0   = blockIdx.x * 256 + (wave >> 1) * 64;
    const float* inb = in + (size_t)b * 512 * 1024;

    v8f acc[2][4];
#pragma unroll
    for (int mi = 0; mi < 2; ++mi)
#pragma unroll
        for (int ni = 0; ni < 4; ++ni) acc[mi][ni] = {};

    for (int c0 = 0; c0 < 512; c0 += 32) {
        v16bf a0 = a_frag_f32(W + (size_t)o0 * 512 + c0, 512);
        v16bf a1 = a_frag_f32(W + (size_t)(o0 + 16) * 512 + c0, 512);
#pragma unroll
        for (int ni = 0; ni < 4; ++ni) {
            v16bf bf = b_frag_f32_krows(inb + (size_t)c0 * 1024 + n0 + ni * 16, 1024);
            acc[0][ni] = wmma_bf16(a0, bf, acc[0][ni]);
            acc[1][ni] = wmma_bf16(a1, bf, acc[1][ni]);
        }
    }

    const int hi = lane >> 4, col = lane & 15;
#pragma unroll
    for (int mi = 0; mi < 2; ++mi)
#pragma unroll
        for (int ni = 0; ni < 4; ++ni)
#pragma unroll
            for (int r = 0; r < 8; ++r) {
                const int o = o0 + mi * 16 + r + 8 * hi;
                const int n = n0 + ni * 16 + col;
                const float val = acc[mi][ni][r] + bias[o];
                const int h = o >> 6, d = o & 63;
                size_t idx;
                if (MODE == 2)
                    idx = ((((size_t)b * 8 + h) * 64 + d) * 1024 + n);
                else
                    idx = ((((size_t)b * 8 + h) * 1024 + n) * 64 + d);
                out[idx] = (__bf16)val;
            }
}

// ---------------------------------------------------------------------------
// Kernel 2: flash attention per (b,h).  q,k: [b][h][1024][64], v: [b][h][64][1024]
// Each wave owns 16 query rows; online softmax over j in tiles of 32.
// ao (attention out) written as [b][h*64+d][n] for the final projection.
// Grid (8 i-tiles of 128, 8 heads, 8 batches), block 256 (8 waves).
// ---------------------------------------------------------------------------
__global__ __launch_bounds__(256) void attn_kernel(
    const __bf16* __restrict__ q, const __bf16* __restrict__ k,
    const __bf16* __restrict__ v, __bf16* __restrict__ ao)
{
    __shared__ __bf16 plds[8 * 16 * 32];  // per-wave 16x32 P repack tile

    const int lane = threadIdx.x & 31;
    const int wave = threadIdx.x >> 5;
    const int b = blockIdx.z, h = blockIdx.y;
    const int i0 = blockIdx.x * 128 + wave * 16;
    const size_t bh = (size_t)b * 8 + h;
    const __bf16* qh = q + bh * 1024 * 64;
    const __bf16* kh = k + bh * 1024 * 64;
    const __bf16* vh = v + bh * 64 * 1024;
    const float scale = 0.125f;  // 64^-0.5, folded into Q (as in reference)

    // Q A-fragments, pre-scaled, reused for every j-tile.
    const v16bf aq0 = a_frag_bf16_scaled(qh + (size_t)i0 * 64, 64, scale);
    const v16bf aq1 = a_frag_bf16_scaled(qh + (size_t)i0 * 64 + 32, 64, scale);

    v8f O[4];
#pragma unroll
    for (int t = 0; t < 4; ++t) O[t] = {};
    float m_run[8], l_run[8];
#pragma unroll
    for (int r = 0; r < 8; ++r) { m_run[r] = -3.0e38f; l_run[r] = 0.0f; }

    __bf16* pl = plds + wave * (16 * 32);
    const int hi = lane >> 4, col = lane & 15;

    for (int j0 = 0; j0 < 1024; j0 += 32) {
        // prefetch next j-tile of K and V (global_prefetch_b8; L2/near hint)
        if (j0 + 32 < 1024) {
            __builtin_prefetch(kh + (size_t)(j0 + 32) * 64 + (size_t)lane * 8, 0, 3);
            __builtin_prefetch(vh + (size_t)(lane & 15) * 1024 + j0 + 32, 0, 3);
        }

        // ---- S tiles: 16 x 32 of logits ----
        v8f s0 = {}, s1 = {};
        {
            v16bf bk;
            bk = b_frag_bf16_ncols(kh + (size_t)j0 * 64, 64);        // d 0..31
            s0 = wmma_bf16(aq0, bk, s0);
            bk = b_frag_bf16_ncols(kh + (size_t)j0 * 64 + 32, 64);   // d 32..63
            s0 = wmma_bf16(aq1, bk, s0);
            bk = b_frag_bf16_ncols(kh + (size_t)(j0 + 16) * 64, 64);
            s1 = wmma_bf16(aq0, bk, s1);
            bk = b_frag_bf16_ncols(kh + (size_t)(j0 + 16) * 64 + 32, 64);
            s1 = wmma_bf16(aq1, bk, s1);
        }

        // ---- online softmax (row = r + 8*hi; 16 cols live in the 16-lane group)
        float alpha[8];
#pragma unroll
        for (int r = 0; r < 8; ++r) {
            const float x0 = s0[r];
            const float x1 = s1[r];
            float mloc = fmaxf(x0, x1);
#pragma unroll
            for (int off = 1; off < 16; off <<= 1)
                mloc = fmaxf(mloc, __shfl_xor(mloc, off, 32));
            const float mnew = fmaxf(m_run[r], mloc);
            alpha[r] = __expf(m_run[r] - mnew);
            const float e0 = __expf(x0 - mnew);
            const float e1 = __expf(x1 - mnew);
            float rs = e0 + e1;
#pragma unroll
            for (int off = 1; off < 16; off <<= 1)
                rs += __shfl_xor(rs, off, 32);
            l_run[r] = l_run[r] * alpha[r] + rs;
            m_run[r] = mnew;
            // stash P (C-layout -> LDS row-major 16x32) for A-fragment repack
            pl[(r + 8 * hi) * 32 + col]      = (__bf16)e0;
            pl[(r + 8 * hi) * 32 + 16 + col] = (__bf16)e1;
        }
#pragma unroll
        for (int t = 0; t < 4; ++t)
#pragma unroll
            for (int r = 0; r < 8; ++r) O[t][r] *= alpha[r];

        // same-wave DS ordering before cross-lane readback
        asm volatile("s_wait_dscnt 0" ::: "memory");
        const v16bf ap = a_frag_bf16(pl, 32);

        // ---- O += P(16x32_j) * V(32_j x 64_d) ----
#pragma unroll
        for (int t = 0; t < 4; ++t) {
            v16bf bv = b_frag_bf16_ncols(vh + (size_t)(t * 16) * 1024 + j0, 1024);
            O[t] = wmma_bf16(ap, bv, O[t]);
        }
    }

    // ---- normalize and store as [b][h*64+d][n] ----
#pragma unroll
    for (int t = 0; t < 4; ++t)
#pragma unroll
        for (int r = 0; r < 8; ++r) {
            const float val = O[t][r] / l_run[r];
            const int i = i0 + r + 8 * hi;
            const int d = t * 16 + col;
            ao[(bh * 64 + d) * 1024 + i] = (__bf16)val;
        }
}

// ---------------------------------------------------------------------------
// Kernel 3: output projection. out[b][o][n] = sum_i Wo[o][i]*ao[b][i][n] + bo[o]
// Wave tile 32(o) x 64(n); block tile 64 x 256; grid (8, 8, 8).
// ---------------------------------------------------------------------------
__global__ __launch_bounds__(256) void proj_out_kernel(
    const __bf16* __restrict__ ao,   // [b][512][1024] bf16
    const float* __restrict__ Wo,    // [512][512]
    const float* __restrict__ bo,    // [512]
    float* __restrict__ out)         // [b][512][1024] f32
{
    const int lane = threadIdx.x & 31;
    const int wave = threadIdx.x >> 5;
    const int b    = blockIdx.z;
    const int o0   = blockIdx.y * 64 + (wave & 1) * 32;
    const int n0   = blockIdx.x * 256 + (wave >> 1) * 64;
    const __bf16* aob = ao + (size_t)b * 512 * 1024;

    v8f acc[2][4];
#pragma unroll
    for (int mi = 0; mi < 2; ++mi)
#pragma unroll
        for (int ni = 0; ni < 4; ++ni) acc[mi][ni] = {};

    for (int c0 = 0; c0 < 512; c0 += 32) {
        v16bf a0 = a_frag_f32(Wo + (size_t)o0 * 512 + c0, 512);
        v16bf a1 = a_frag_f32(Wo + (size_t)(o0 + 16) * 512 + c0, 512);
#pragma unroll
        for (int ni = 0; ni < 4; ++ni) {
            v16bf bf = b_frag_bf16_krows(aob + (size_t)c0 * 1024 + n0 + ni * 16, 1024);
            acc[0][ni] = wmma_bf16(a0, bf, acc[0][ni]);
            acc[1][ni] = wmma_bf16(a1, bf, acc[1][ni]);
        }
    }

    const int hi = lane >> 4, col = lane & 15;
#pragma unroll
    for (int mi = 0; mi < 2; ++mi)
#pragma unroll
        for (int ni = 0; ni < 4; ++ni)
#pragma unroll
            for (int r = 0; r < 8; ++r) {
                const int o = o0 + mi * 16 + r + 8 * hi;
                const int n = n0 + ni * 16 + col;
                out[((size_t)b * 512 + o) * 1024 + n] = acc[mi][ni][r] + bo[o];
            }
}

// ---------------------------------------------------------------------------
extern "C" void kernel_launch(void* const* d_in, const int* in_sizes, int n_in,
                              void* d_out, int out_size, void* d_ws, size_t ws_size,
                              hipStream_t stream) {
    (void)in_sizes; (void)n_in; (void)out_size; (void)ws_size;

    const float* x   = (const float*)d_in[0];
    const float* ctx = (const float*)d_in[1];
    const float* Wq  = (const float*)d_in[2];
    const float* bq  = (const float*)d_in[3];
    const float* Wk  = (const float*)d_in[4];
    const float* bk  = (const float*)d_in[5];
    const float* Wv  = (const float*)d_in[6];
    const float* bv  = (const float*)d_in[7];
    const float* Wo  = (const float*)d_in[8];
    const float* bo  = (const float*)d_in[9];
    float* out = (float*)d_out;

    // workspace: 4 bf16 segments of 8*8*1024*64 elements (8 MiB each, 32 MiB total)
    const size_t SEG = (size_t)8 * 8 * 1024 * 64;
    __bf16* qws  = (__bf16*)d_ws;
    __bf16* kws  = qws + SEG;
    __bf16* vws  = kws + SEG;
    __bf16* aows = vws + SEG;

    const dim3 blk(256);
    const dim3 grd_proj(4, 8, 8);
    const dim3 grd_attn(8, 8, 8);

    proj_qkv_kernel<0><<<grd_proj, blk, 0, stream>>>(x,   Wq, bq, qws);
    proj_qkv_kernel<0><<<grd_proj, blk, 0, stream>>>(ctx, Wk, bk, kws);
    proj_qkv_kernel<2><<<grd_proj, blk, 0, stream>>>(ctx, Wv, bv, vws);
    attn_kernel<<<grd_attn, blk, 0, stream>>>(qws, kws, vws, aows);
    proj_out_kernel<<<grd_attn, blk, 0, stream>>>(aows, Wo, bo, out);
}
